// OutlookTransformerBlock_82437602280005
// MI455X (gfx1250) — compile-verified, matrix operand order checked by reading, not run on hardware
//
#include <hip/hip_runtime.h>
#include <hip/hip_bf16.h>

// ---------------- problem constants ----------------
constexpr int kB     = 4;
constexpr int kC     = 192;
constexpr int kHEADS = 6;
constexpr int kDH    = 32;       // head dim
constexpr int kHW    = 9216;     // 96*96
constexpr int kNT    = kB * kHW; // 36864 total pixels
constexpr int kCA    = 486;      // attention channels
constexpr int kE     = 384;      // expansion channels
constexpr int kIMG   = 96;

typedef __attribute__((ext_vector_type(16))) _Float16 v16h;
typedef __attribute__((ext_vector_type(8)))  float    v8f;

// CDNA5 async global->LDS path (ASYNCcnt), guarded so compilation can't break
#if defined(__has_builtin)
#if __has_builtin(__builtin_amdgcn_global_load_async_to_lds_b128) && \
    __has_builtin(__builtin_amdgcn_s_wait_asynccnt)
#define USE_ASYNC_LDS 1
#endif
#endif
#ifndef USE_ASYNC_LDS
#define USE_ASYNC_LDS 0
#endif

#if USE_ASYNC_LDS
// builtin expects: (global int4*, shared int4*, imm offset, imm cpol)
typedef int v4i_b __attribute__((vector_size(16)));
typedef __attribute__((address_space(1))) v4i_b* gasync_ptr;
typedef __attribute__((address_space(3))) v4i_b* lasync_ptr;
#endif

// f16 activation tensors use "X8" layout: [row/8][col][8 halves]
// -> 8 consecutive k per (col) are contiguous (16B), so GEMM staging and
//    WMMA fragment builds are pure 128-bit copies.
__device__ __forceinline__ size_t x8idx(int row, int col) {
    return ((size_t)(row >> 3) * kNT + (size_t)col) * 8 + (row & 7);
}

// ============================================================
// LayerNorm over channels -> f16 X8 output.
// addr(c, n) = n + b*bmul + c*cstride   (b = n / HW)
//   NCHW input : cstride = HW,  bmul = (C-1)*HW
//   CP   input : cstride = NT,  bmul = 0
// ============================================================
__global__ __launch_bounds__(256) void ln_to_f16_kernel(
    const float* __restrict__ x, const float* __restrict__ g,
    const float* __restrict__ be, _Float16* __restrict__ out,
    int cstride, int bmul, float eps)
{
    __shared__ float s_sum[8][32];
    __shared__ float s_sq[8][32];
    __shared__ float s_mean[32], s_rstd[32];
    const int tx = threadIdx.x, ty = threadIdx.y;
    const int n  = blockIdx.x * 32 + tx;
    const int b  = n / kHW;
    const size_t base = (size_t)n + (size_t)b * (size_t)bmul;

    float vals[kC / 8];
    float sum = 0.f, sq = 0.f;
#pragma unroll
    for (int i = 0; i < kC / 8; ++i) {
        int c = ty + i * 8;
        float v = x[base + (size_t)c * (size_t)cstride];
        vals[i] = v; sum += v; sq += v * v;
    }
    s_sum[ty][tx] = sum; s_sq[ty][tx] = sq;
    __syncthreads();
    if (ty == 0) {
        float s = 0.f, q = 0.f;
#pragma unroll
        for (int j = 0; j < 8; ++j) { s += s_sum[j][tx]; q += s_sq[j][tx]; }
        float mean = s / (float)kC;
        float var  = q / (float)kC - mean * mean;
        s_mean[tx] = mean;
        s_rstd[tx] = rsqrtf(var + eps);
    }
    __syncthreads();
    const float mean = s_mean[tx], rstd = s_rstd[tx];
#pragma unroll
    for (int i = 0; i < kC / 8; ++i) {
        int c = ty + i * 8;   // c>>3 == i, c&7 == ty
        float v = (vals[i] - mean) * rstd * g[c] + be[c];
        out[((size_t)i * kNT + n) * 8 + ty] = (_Float16)v;
    }
}

// ============================================================
// WMMA GEMM: Out[M, NT] = W[M,K](f32->f16) x X[K,NT](f16 X8) + bias
// Block 64x128, 8 waves, wave = 32x32 (2x2 WMMA 16x16x32 tiles).
// Epilogues:
//  0: f32 CP out                 1: SiLU -> f32 CP out
//  2: + res(NCHW f32) -> f32 CP  3: res(CP f32) + SiLU -> f16 X8
//  4: + res(CP f32) -> f32 NCHW out
// ============================================================
__global__ __launch_bounds__(256) void gemm_wmma_kernel(
    const _Float16* __restrict__ X, const float* __restrict__ W,
    const float* __restrict__ bias, const float* __restrict__ res,
    float* __restrict__ out_f, _Float16* __restrict__ out_h,
    int M, int K, int epi)
{
    // both tiles k-contiguous per row/col; pitch 40 halves = 80B (16B mult)
    __shared__ __align__(16) _Float16 As[64][40];    // [m][k]
    __shared__ __align__(16) _Float16 Bt[128][40];   // [n][k]

    const int tid  = threadIdx.x;
    const int lane = tid & 31;
    const int wave = tid >> 5;
    const int wm = wave & 1;        // M half    (0..1)
    const int wn = wave >> 1;       // N quarter (0..3)
    const int mbase = blockIdx.y * 64;
    const int nbase = blockIdx.x * 128;

    v8f acc[2][2] = {};

    const int ar = tid >> 2;            // A row 0..63
    const int ak = (tid & 3) * 8;       // 8 consecutive k
    const int bkg = tid >> 6;           // B k-group 0..3 (8 k each)
    const int bc  = (tid & 63) * 2;     // 2 consecutive col-chunks

    const int lm = lane & 15;
    const int hi = (lane >> 4) << 3;    // +8 k-offset for upper half-wave

    const int arow_g = mbase + ar;
    const bool arow_ok = arow_g < M;

    union AFrag { v16h v; uint4 u[2]; };

    for (int k0 = 0; k0 < K; k0 += 32) {
        // ---- stage A tile: 2x float4 load, convert, 1x b128 LDS store ----
        {
            float4 w0 = make_float4(0.f, 0.f, 0.f, 0.f);
            float4 w1 = make_float4(0.f, 0.f, 0.f, 0.f);
            if (arow_ok) {
                const float4* wp = (const float4*)(W + (size_t)arow_g * K + k0 + ak);
                w0 = wp[0];
                w1 = wp[1];
            }
            union { _Float16 h[8]; uint4 u; } pk;
            pk.h[0] = (_Float16)w0.x; pk.h[1] = (_Float16)w0.y;
            pk.h[2] = (_Float16)w0.z; pk.h[3] = (_Float16)w0.w;
            pk.h[4] = (_Float16)w1.x; pk.h[5] = (_Float16)w1.y;
            pk.h[6] = (_Float16)w1.z; pk.h[7] = (_Float16)w1.w;
            *(uint4*)&As[ar][ak] = pk.u;
        }
        // ---- stage B tile: f16 X8 layout, no conversion needed ----
        {
            const _Float16* g0 = X + ((size_t)(k0 / 8 + bkg) * kNT + nbase + bc) * 8;
#if USE_ASYNC_LDS
            // CDNA5 async DMA: global -> LDS without VGPR round trip (ASYNCcnt)
            __builtin_amdgcn_global_load_async_to_lds_b128(
                (gasync_ptr)g0, (lasync_ptr)&Bt[bc][bkg * 8], 0, 0);
            __builtin_amdgcn_global_load_async_to_lds_b128(
                (gasync_ptr)(g0 + 8), (lasync_ptr)&Bt[bc + 1][bkg * 8], 0, 0);
#else
            const uint4* src = (const uint4*)g0;
            *(uint4*)&Bt[bc][bkg * 8]     = src[0];
            *(uint4*)&Bt[bc + 1][bkg * 8] = src[1];
#endif
        }
#if USE_ASYNC_LDS
        __builtin_amdgcn_s_wait_asynccnt(0);
#endif
        __syncthreads();

        // ---- fragments: 2x ds_load_b128 each ----
        AFrag afr[2], bfr[2];
#pragma unroll
        for (int ms = 0; ms < 2; ++ms) {
            const int row = wm * 32 + ms * 16 + lm;
            afr[ms].u[0] = *(const uint4*)&As[row][hi];       // halves 0..7  : k=hi..hi+7
            afr[ms].u[1] = *(const uint4*)&As[row][hi + 16];  // halves 8..15 : k=hi+16..hi+23
        }
#pragma unroll
        for (int ns = 0; ns < 2; ++ns) {
            const int col = wn * 32 + ns * 16 + lm;
            bfr[ns].u[0] = *(const uint4*)&Bt[col][hi];
            bfr[ns].u[1] = *(const uint4*)&Bt[col][hi + 16];
        }
#pragma unroll
        for (int ms = 0; ms < 2; ++ms)
#pragma unroll
            for (int ns = 0; ns < 2; ++ns)
                acc[ms][ns] = __builtin_amdgcn_wmma_f32_16x16x32_f16(
                    false, afr[ms].v, false, bfr[ns].v,
                    (short)0, acc[ms][ns], false, false);
        __syncthreads();
    }

    // epilogue (C/D layout: lane -> col lm, vgpr r -> row r + 8*(lane>=16))
    const int rsel = (lane >> 4) << 3;
    const int bb   = nbase / kHW;           // tile never crosses batch (128 | 9216)
    const int pbas = nbase - bb * kHW;
#pragma unroll
    for (int ms = 0; ms < 2; ++ms) {
#pragma unroll
        for (int ns = 0; ns < 2; ++ns) {
            const int coff = wn * 32 + ns * 16 + lm;
            const int col  = nbase + coff;
#pragma unroll
            for (int r = 0; r < 8; ++r) {
                int row = mbase + wm * 32 + ms * 16 + r + rsel;
                if (row >= M) continue;
                float v = acc[ms][ns][r] + bias[row];
                size_t cp = (size_t)row * kNT + col;
                if (epi == 0) {
                    out_f[cp] = v;
                } else if (epi == 1) {
                    out_f[cp] = v / (1.f + __expf(-v));
                } else if (epi == 2) {
                    v += res[(size_t)bb * kC * kHW + (size_t)row * kHW + (pbas + coff)];
                    out_f[cp] = v;
                } else if (epi == 3) {
                    float s = v / (1.f + __expf(-v));
                    out_h[x8idx(row, col)] = (_Float16)(res[cp] + s);
                } else { // 4: final residual, NCHW f32 output
                    v += res[cp];
                    out_f[(size_t)bb * kC * kHW + (size_t)row * kHW + (pbas + coff)] = v;
                }
            }
        }
    }
}

// ============================================================
// Outlook attention: softmax(sim*scale) @ unfolded(v), fold via
// coalesced global f32 atomics into pre-zeroed folded[C, NT].
// Block: 32 pixels x 1 head, (32,8) threads.
// ============================================================
__global__ __launch_bounds__(256) void outlook_attn_kernel(
    const float* __restrict__ sim,  // [CA, NT] CP layout
    const float* __restrict__ vv,   // [C, NT]  CP layout
    float* __restrict__ folded)     // [C, NT]  zeroed
{
    __shared__ float att[32][84];
    const int tx = threadIdx.x, ty = threadIdx.y;
    const int head = blockIdx.y;
    const int nb = blockIdx.x * 32;
    const int b  = nb / kHW;
    const int p  = (nb - b * kHW) + tx;       // in-batch pixel
    const int ph = p / kIMG, pw = p - ph * kIMG;
    const float scale = 0.17677669529663687f; // 32^-0.5

    // phase 1: load sim with the reference's flat reinterpretation
    for (int q = ty; q < 81; q += 8) {
        int flat = p * kCA + head * 81 + q;   // flat within batch, (486,HW) layout
        int cc  = flat / kHW;
        int pix = flat - cc * kHW;
        att[tx][q] = sim[(size_t)cc * kNT + (size_t)b * kHW + pix] * scale;
    }
    __syncthreads();

    // phase 2: softmax over j for each i (ty=0 also handles i=8)
    for (int i = ty; i < 9; i += 8) {
        float m = -1e30f;
#pragma unroll
        for (int j = 0; j < 9; ++j) m = fmaxf(m, att[tx][i * 9 + j]);
        float e[9], s = 0.f;
#pragma unroll
        for (int j = 0; j < 9; ++j) { e[j] = __expf(att[tx][i * 9 + j] - m); s += e[j]; }
        float inv = 1.f / s;
#pragma unroll
        for (int j = 0; j < 9; ++j) att[tx][i * 9 + j] = e[j] * inv;
    }
    __syncthreads();

    // phase 3: gather value patches (9x4 per thread), attn matmul, fold scatter
    const int dd0 = ty * 4;
    float vp[9][4];
#pragma unroll
    for (int u = 0; u < 9; ++u) {
        int qh = ph + u / 3 - 1;
        int qw = pw + u % 3 - 1;
        bool ok = (qh >= 0 && qh < kIMG && qw >= 0 && qw < kIMG);
        size_t ncol = (size_t)b * kHW + (size_t)(ok ? qh * kIMG + qw : 0);
#pragma unroll
        for (int dd = 0; dd < 4; ++dd)
            vp[u][dd] = ok ? vv[(size_t)(head * kDH + dd0 + dd) * kNT + ncol] : 0.f;
    }
#pragma unroll
    for (int e9 = 0; e9 < 9; ++e9) {
        int th = ph + e9 / 3 - 1;
        int tw = pw + e9 % 3 - 1;
        if (th < 0 || th >= kIMG || tw < 0 || tw >= kIMG) continue;
        float acc[4] = {0.f, 0.f, 0.f, 0.f};
#pragma unroll
        for (int u = 0; u < 9; ++u) {
            float a = att[tx][e9 * 9 + u];
#pragma unroll
            for (int dd = 0; dd < 4; ++dd) acc[dd] += a * vp[u][dd];
        }
        size_t tcol = (size_t)b * kHW + th * kIMG + tw;
#pragma unroll
        for (int dd = 0; dd < 4; ++dd)
            atomicAdd(&folded[(size_t)(head * kDH + dd0 + dd) * kNT + tcol], acc[dd]);
    }
}

// ============================================================
// Fused 3x3 / 5x5 / 7x7 depthwise convs -> f16 X8 concat [3E, NT]
// ============================================================
__global__ __launch_bounds__(128) void dwconv357_kernel(
    const float* __restrict__ y,
    const float* __restrict__ w3, const float* __restrict__ b3,
    const float* __restrict__ w5, const float* __restrict__ b5,
    const float* __restrict__ w7, const float* __restrict__ b7,
    _Float16* __restrict__ cat)
{
    const int c = blockIdx.y;
    const int n = blockIdx.x * 128 + threadIdx.x;
    const int b = n / kHW;
    const int p = n - b * kHW;
    const int ph = p / kIMG, pw = p - ph * kIMG;
    const float* yc = y + (size_t)c * kNT + (size_t)b * kHW;
    float a3 = b3[c], a5 = b5[c], a7 = b7[c];
#pragma unroll
    for (int dy = -3; dy <= 3; ++dy) {
        int qh = ph + dy;
        if (qh < 0 || qh >= kIMG) continue;
#pragma unroll
        for (int dx = -3; dx <= 3; ++dx) {
            int qw = pw + dx;
            if (qw < 0 || qw >= kIMG) continue;
            float v = yc[qh * kIMG + qw];
            a7 += v * w7[c * 49 + (dy + 3) * 7 + (dx + 3)];
            if (dy >= -2 && dy <= 2 && dx >= -2 && dx <= 2)
                a5 += v * w5[c * 25 + (dy + 2) * 5 + (dx + 2)];
            if (dy >= -1 && dy <= 1 && dx >= -1 && dx <= 1)
                a3 += v * w3[c * 9 + (dy + 1) * 3 + (dx + 1)];
        }
    }
    cat[x8idx(c, n)]          = (_Float16)a3;
    cat[x8idx(kE + c, n)]     = (_Float16)a5;
    cat[x8idx(2 * kE + c, n)] = (_Float16)a7;
}

__global__ void zero_f32_kernel(float* __restrict__ p, size_t n) {
    size_t i = (size_t)blockIdx.x * blockDim.x + threadIdx.x;
    size_t st = (size_t)gridDim.x * blockDim.x;
    for (; i < n; i += st) p[i] = 0.f;
}

// f32 CP [rows, NT] -> f16 X8
__global__ void f32cp_to_f16x8_kernel(const float* __restrict__ in,
                                      _Float16* __restrict__ out, size_t total) {
    size_t i = (size_t)blockIdx.x * blockDim.x + threadIdx.x;
    size_t st = (size_t)gridDim.x * blockDim.x;
    for (; i < total; i += st) {
        int row = (int)(i / kNT);
        int col = (int)(i - (size_t)row * kNT);
        out[x8idx(row, col)] = (_Float16)in[i];
    }
}

// ============================================================
extern "C" void kernel_launch(void* const* d_in, const int* in_sizes, int n_in,
                              void* d_out, int out_size, void* d_ws, size_t ws_size,
                              hipStream_t stream) {
    (void)in_sizes; (void)n_in; (void)out_size; (void)ws_size;
    const float* feats = (const float*)d_in[0];
    const float* ln1_g = (const float*)d_in[1];
    const float* ln1_b = (const float*)d_in[2];
    const float* Wv    = (const float*)d_in[3];
    const float* bv    = (const float*)d_in[4];
    const float* Wa    = (const float*)d_in[5];
    const float* ba    = (const float*)d_in[6];
    const float* Wo    = (const float*)d_in[7];
    const float* bo    = (const float*)d_in[8];
    const float* ln2_g = (const float*)d_in[9];
    const float* ln2_b = (const float*)d_in[10];
    const float* Win   = (const float*)d_in[11];
    const float* bin_w = (const float*)d_in[12];
    const float* Wd3   = (const float*)d_in[13];
    const float* bd3   = (const float*)d_in[14];
    const float* Wd5   = (const float*)d_in[15];
    const float* bd5   = (const float*)d_in[16];
    const float* Wd7   = (const float*)d_in[17];
    const float* bd7   = (const float*)d_in[18];
    const float* Wpw   = (const float*)d_in[19];
    const float* bpw   = (const float*)d_in[20];
    const float* Wout  = (const float*)d_in[21];
    const float* bout  = (const float*)d_in[22];

    char* ws = (char*)d_ws;
    size_t off = 0;
    auto alloc = [&](size_t bytes) -> void* {
        void* pp = ws + off;
        off = (off + bytes + 255) & ~(size_t)255;
        return pp;
    };
    _Float16* xn1    = (_Float16*)alloc((size_t)kC * kNT * 2);
    float*    v_f    = (float*)   alloc((size_t)kC * kNT * 4);
    float*    sim_f  = (float*)   alloc((size_t)kCA * kNT * 4);
    float*    fold_f = (float*)   alloc((size_t)kC * kNT * 4);
    _Float16* fold_h = (_Float16*)alloc((size_t)kC * kNT * 2);
    float*    feats1 = (float*)   alloc((size_t)kC * kNT * 4);
    _Float16* xn2    = (_Float16*)alloc((size_t)kC * kNT * 2);
    float*    y1_f   = (float*)   alloc((size_t)kE * kNT * 4);
    _Float16* cat_h  = (_Float16*)alloc((size_t)3 * kE * kNT * 2);
    _Float16* y_h    = (_Float16*)alloc((size_t)kE * kNT * 2);

    const dim3 lnBlk(32, 8);
    const int  nTileBlocks = kNT / 128;  // 288

    // ---- outlook attention branch ----
    ln_to_f16_kernel<<<kNT / 32, lnBlk, 0, stream>>>(
        feats, ln1_g, ln1_b, xn1, kHW, (kC - 1) * kHW, 1e-6f);
    gemm_wmma_kernel<<<dim3(nTileBlocks, 3), 256, 0, stream>>>(
        xn1, Wv, bv, nullptr, v_f, nullptr, kC, kC, 0);
    gemm_wmma_kernel<<<dim3(nTileBlocks, 8), 256, 0, stream>>>(
        xn1, Wa, ba, nullptr, sim_f, nullptr, kCA, kC, 0);
    zero_f32_kernel<<<2048, 256, 0, stream>>>(fold_f, (size_t)kC * kNT);
    outlook_attn_kernel<<<dim3(kNT / 32, kHEADS), lnBlk, 0, stream>>>(
        sim_f, v_f, fold_f);
    f32cp_to_f16x8_kernel<<<2048, 256, 0, stream>>>(
        fold_f, fold_h, (size_t)kC * kNT);
    gemm_wmma_kernel<<<dim3(nTileBlocks, 3), 256, 0, stream>>>(
        fold_h, Wo, bo, feats, feats1, nullptr, kC, kC, 2);   // + residual(NCHW)

    // ---- feedforward branch ----
    ln_to_f16_kernel<<<kNT / 32, lnBlk, 0, stream>>>(
        feats1, ln2_g, ln2_b, xn2, kNT, 0, 1e-5f);
    gemm_wmma_kernel<<<dim3(nTileBlocks, 6), 256, 0, stream>>>(
        xn2, Win, bin_w, nullptr, y1_f, nullptr, kE, kC, 1);  // SiLU
    dwconv357_kernel<<<dim3(nTileBlocks, kE), 128, 0, stream>>>(
        y1_f, Wd3, bd3, Wd5, bd5, Wd7, bd7, cat_h);
    gemm_wmma_kernel<<<dim3(nTileBlocks, 6), 256, 0, stream>>>(
        cat_h, Wpw, bpw, y1_f, nullptr, y_h, kE, 3 * kE, 3);  // y1 + SiLU -> f16
    gemm_wmma_kernel<<<dim3(nTileBlocks, 3), 256, 0, stream>>>(
        y_h, Wout, bout, feats1, (float*)d_out, nullptr, kC, kE, 4); // final NCHW
}
